// Attention_57853209477370
// MI455X (gfx1250) — matrix operand order repaired and checked by reading, not compiled, not run on hardware
//
#include <hip/hip_runtime.h>
#include <hip/hip_bf16.h>

typedef __bf16 bf16;
typedef __attribute__((ext_vector_type(16))) __bf16 v16bf;
typedef __attribute__((ext_vector_type(8)))  float  v8f;
typedef __attribute__((ext_vector_type(4)))  unsigned int u32x4;
typedef __attribute__((ext_vector_type(8)))  int i32x8;
typedef __attribute__((ext_vector_type(4)))  int i32x4;

#define WMMA_BF16(a, b, c) \
  __builtin_amdgcn_wmma_f32_16x16x32_bf16(false, (a), false, (b), (short)0, (c), false, false)

#if defined(__has_builtin)
#if __has_builtin(__builtin_amdgcn_tensor_load_to_lds)
#define HAVE_TDM 1
#endif
#endif

// ---------------- fragment loaders (ISA 7.12.2 layouts) ----------------
// A 16x32 bf16, row-major [m][k]: lane l (ln=l&15, lh=l>>4) holds row m+ln,
// elements 0..7 at k+lh*8, elements 8..15 at k+16+lh*8 (two 16B chunks).
static __device__ __forceinline__ v16bf frag_a(const bf16* base, int pitch, int m, int k, int lane) {
  const int ln = lane & 15, lh = lane >> 4;
  const bf16* r = base + (size_t)(m + ln) * pitch + k + lh * 8;
  v16bf a;
  ((uint4*)&a)[0] = *(const uint4*)(r);
  ((uint4*)&a)[1] = *(const uint4*)(r + 16);
  return a;
}
// B 32x16 bf16 stored transposed [n][k]: lane holds col n+ln,
// elements 0..7 at k+lh*8, elements 8..15 at k+16+lh*8 (two 16B chunks).
static __device__ __forceinline__ v16bf frag_b(const bf16* base, int pitch, int n, int k, int lane) {
  const int ln = lane & 15, lh = lane >> 4;
  const bf16* r = base + (size_t)(n + ln) * pitch + k + lh * 8;
  v16bf b;
  ((uint4*)&b)[0] = *(const uint4*)(r);
  ((uint4*)&b)[1] = *(const uint4*)(r + 16);
  return b;
}

// ---------------- TDM 2D tile load: global (strided) -> LDS (compact) -----
// D# per ISA 08_async_tensor §8.3/8.4: group0 = {count/type, lds_addr,
// global_addr}, group1 = {data_size=2B, tensor dims, tile dims, stride}.
// Rows past dim1_valid are zero-filled by the TDM (OOB reads return zero).
#if HAVE_TDM
static __device__ __forceinline__ void tdm_load_2d(unsigned lds_byte_off, const void* gptr,
                                                   unsigned dim0, unsigned dim1_valid,
                                                   unsigned tile0, unsigned tile1,
                                                   unsigned stride_elems) {
  unsigned long long ga = (unsigned long long)gptr;
  u32x4 g0;
  g0.x = 0x1u;                                             // count=1 (valid), user mode
  g0.y = lds_byte_off;                                     // lds_addr [63:32]
  g0.z = (unsigned)(ga & 0xffffffffu);                     // global_addr low
  g0.w = (unsigned)((ga >> 32) & 0x01ffffffu) | (2u << 30);// global_addr high | type=2
  i32x8 g1;
  g1[0] = (int)(1u << 16);                                 // data_size=1 -> 2 bytes
  g1[1] = (int)((dim0 & 0xffffu) << 16);                   // tensor_dim0[15:0] @ bits 63:48
  g1[2] = (int)((dim0 >> 16) | ((dim1_valid & 0xffffu) << 16)); // dim0 hi | dim1 lo
  g1[3] = (int)((dim1_valid >> 16) | ((tile0 & 0xffffu) << 16));// dim1 hi | tile_dim0
  g1[4] = (int)(tile1 & 0xffffu);                          // tile_dim1 (tile_dim2=0)
  g1[5] = (int)stride_elems;                               // tensor_dim0_stride low32
  g1[6] = 0;                                               // stride hi | dim1_stride lo
  g1[7] = 0;
  i32x4 z4 = {0, 0, 0, 0};
#if __clang_major__ >= 23
  i32x8 z8 = {0, 0, 0, 0, 0, 0, 0, 0};
  __builtin_amdgcn_tensor_load_to_lds(g0, g1, z4, z4, z8, 0);
#else
  __builtin_amdgcn_tensor_load_to_lds(g0, g1, z4, z4, 0);
#endif
}
#endif

// ---------------- prep kernels ----------------
__global__ __launch_bounds__(256) void cvt_f32_bf16(const float* __restrict__ in,
                                                    bf16* __restrict__ out, int n) {
  int i = blockIdx.x * 256 + threadIdx.x;
  if (i < n) out[i] = (bf16)in[i];
}

__global__ __launch_bounds__(256) void copy_f32(const float* __restrict__ in,
                                                float* __restrict__ out, int n) {
  int i = blockIdx.x * 256 + threadIdx.x;
  if (i < n) out[i] = in[i];
}

// W_embed (2048x768) -> bf16 transposed [768][2048]
__global__ __launch_bounds__(256) void transpose_wembed(const float* __restrict__ in,
                                                        bf16* __restrict__ out) {
  int i = blockIdx.x * 256 + threadIdx.x;
  if (i >= 768 * 2048) return;
  int n = i / 2048, k = i % 2048;
  out[i] = (bf16)in[(size_t)k * 768 + n];
}

// protoT_pad[c][32][768]: rows 0..19 = dup_proto[c][:, p]^T, row 20 = dup_pooling[c], rest 0
__global__ __launch_bounds__(256) void pack_proto(const float* __restrict__ proto,
                                                  const float* __restrict__ pooling,
                                                  bf16* __restrict__ out) {
  int i = blockIdx.x * 256 + threadIdx.x;
  if (i >= 1000 * 32 * 768) return;
  int c = i / (32 * 768);
  int rem = i - c * (32 * 768);
  int p = rem / 768, d = rem - p * 768;
  float v = 0.f;
  if (p < 20)       v = proto[(size_t)c * 768 * 20 + (size_t)d * 20 + p];
  else if (p == 20) v = pooling[(size_t)c * 768 + d];
  out[i] = (bf16)v;
}

// ---------------- generic bf16 WMMA GEMM ----------------
// C[M][N] = act(A[M][K] @ B[N][K]^T + bias). A row-major MK, B row-major NK.
// act: 0=none, 1=relu, 2=leaky(x>0 -> 2x). Optional f32 out (row-major) and
// bf16 out; if groupC>0 the bf16 out is written as [c][b][N] with c=row%groupC.
// A rows past M are CLAMPED (branchless); their outputs are masked at store.
#define APITCH 48
__global__ __launch_bounds__(128)
void gemm_bf16(const bf16* __restrict__ A, const bf16* __restrict__ B,
               const float* __restrict__ bias, float* __restrict__ outF,
               bf16* __restrict__ outH, int groupC,
               int M, int N, int K, int act) {
  __shared__ __align__(16) bf16 Alds[64 * APITCH];
  __shared__ __align__(16) bf16 Blds[64 * APITCH];
  const int tid = threadIdx.x, lane = tid & 31, wave = tid >> 5;
  const int ln = lane & 15, lh = lane >> 4;
  const int mBlock = blockIdx.y * 64, nBlock = blockIdx.x * 64;
  v8f acc[4] = {v8f{}, v8f{}, v8f{}, v8f{}};

  // each thread owns two fixed (row, kchunk) slots of the 64x32 tiles
  const int r0 = tid >> 2, r1 = (tid + 128) >> 2;
  const int kc0 = (tid & 3) << 3, kc1 = kc0;  // chunks 0..3 per row
  int gra0 = mBlock + r0; if (gra0 >= M) gra0 = M - 1;
  int gra1 = mBlock + r1; if (gra1 >= M) gra1 = M - 1;
  const bf16* Ap0 = A + (size_t)gra0 * K + kc0;
  const bf16* Ap1 = A + (size_t)gra1 * K + kc1;
  const bf16* Bp0 = B + (size_t)(nBlock + r0) * K + kc0;
  const bf16* Bp1 = B + (size_t)(nBlock + r1) * K + kc1;

  for (int k0 = 0; k0 < K; k0 += 32) {
    uint4 a0 = *(const uint4*)(Ap0 + k0);
    uint4 a1 = *(const uint4*)(Ap1 + k0);
    uint4 b0 = *(const uint4*)(Bp0 + k0);
    uint4 b1 = *(const uint4*)(Bp1 + k0);
    if (k0 + 32 < K) {  // keep HBM requests ahead of the matrix pipe
      __builtin_prefetch(Ap0 + k0 + 32, 0, 1);
      __builtin_prefetch(Bp0 + k0 + 32, 0, 1);
    }
    *(uint4*)&Alds[r0 * APITCH + kc0] = a0;
    *(uint4*)&Alds[r1 * APITCH + kc1] = a1;
    *(uint4*)&Blds[r0 * APITCH + kc0] = b0;
    *(uint4*)&Blds[r1 * APITCH + kc1] = b1;
    __syncthreads();
    v16bf a = frag_a(Alds, APITCH, wave * 16, 0, lane);
#pragma unroll
    for (int nt = 0; nt < 4; nt++) {
      v16bf bb = frag_b(Blds, APITCH, nt * 16, 0, lane);
      acc[nt] = WMMA_BF16(a, bb, acc[nt]);
    }
    __syncthreads();
  }

  const int rowsB = (groupC > 0) ? (M / groupC) : 0;
#pragma unroll
  for (int nt = 0; nt < 4; nt++) {
    const int col = nBlock + nt * 16 + ln;
    const float bs = bias ? bias[col] : 0.f;
#pragma unroll
    for (int r = 0; r < 8; r++) {
      int row = mBlock + wave * 16 + r + 8 * lh;
      if (row >= M) continue;
      float v = acc[nt][r] + bs;
      if (act == 1)      v = fmaxf(v, 0.f);
      else if (act == 2) v = (v > 0.f) ? v * 2.0f : v;
      if (outF) outF[(size_t)row * N + col] = v;
      if (outH) {
        size_t idx;
        if (groupC > 0) {
          int cc = row % groupC, bb2 = row / groupC;
          idx = ((size_t)cc * rowsB + bb2) * N + col;
        } else idx = (size_t)row * N + col;
        outH[idx] = (bf16)v;
      }
    }
  }
}

// ---------------- attention kernel ----------------
// One block per (64-query chunk, head, batch). scores = q k^T / sqrt(96),
// softmax over S=196, out = attn @ v. q/k tiles DMA'd by the Tensor Data
// Mover (zero-padding OOB rows); all GEMMs via WMMA bf16.
__global__ __launch_bounds__(128)
void attn_kernel(const bf16* __restrict__ q, const bf16* __restrict__ k,
                 const bf16* __restrict__ v, bf16* __restrict__ h_att) {
  constexpr int S = 196;
  __shared__ __align__(16) bf16  q_lds[64 * 96];    // 12 KB
  __shared__ __align__(16) bf16  kv_lds[96 * 224];  // 42 KB: k[s<208][96], then v^T[d][224]
  __shared__ __align__(16) float sc[64 * 208];      // 52 KB
  __shared__ __align__(16) bf16  at[64 * 224];      // 28 KB
  const int c0 = blockIdx.x * 64, hh = blockIdx.y, b = blockIdx.z;
  const int tid = threadIdx.x, lane = tid & 31, wave = tid >> 5;
  const int ln = lane & 15, lh = lane >> 4;
  const int m = wave * 16;

#if HAVE_TDM
  if (tid < 32) {  // one wave issues both DMAs; TENSORcnt is per-wave
    tdm_load_2d((unsigned)(unsigned long long)(void*)q_lds,
                q + (size_t)c0 * 768 + hh * 96,
                /*dim0=*/96, /*dim1_valid=*/(unsigned)(1000 - c0),
                /*tile0=*/96, /*tile1=*/64, /*stride=*/768);
    tdm_load_2d((unsigned)(unsigned long long)(void*)kv_lds,
                k + (size_t)b * S * 768 + hh * 96,
                /*dim0=*/96, /*dim1_valid=*/S,
                /*tile0=*/96, /*tile1=*/208, /*stride=*/768);
    __builtin_amdgcn_s_wait_tensorcnt((short)0);
  }
#else
  for (int ch = tid; ch < 64 * 12; ch += 128) {
    int r = ch / 12, dc = (ch % 12) * 8, c = c0 + r;
    uint4 val = {0, 0, 0, 0};
    if (c < 1000) val = *(const uint4*)(q + (size_t)c * 768 + hh * 96 + dc);
    *(uint4*)&q_lds[r * 96 + dc] = val;
  }
  for (int ch = tid; ch < 208 * 12; ch += 128) {
    int s = ch / 12, dc = (ch % 12) * 8;
    uint4 val = {0, 0, 0, 0};
    if (s < S) val = *(const uint4*)(k + ((size_t)b * S + s) * 768 + hh * 96 + dc);
    *(uint4*)&kv_lds[s * 96 + dc] = val;
  }
#endif
  __syncthreads();

  const float scale = 0.10206207261596577f;  // 1/sqrt(96)
  for (int st = 0; st < 13; ++st) {
    v8f acc = {};
#pragma unroll
    for (int kk = 0; kk < 3; ++kk) {
      v16bf a  = frag_a(q_lds, 96, m, kk * 32, lane);
      v16bf bb = frag_b(kv_lds, 96, st * 16, kk * 32, lane);
      acc = WMMA_BF16(a, bb, acc);
    }
#pragma unroll
    for (int r = 0; r < 8; r++)
      sc[(m + r + 8 * lh) * 208 + st * 16 + ln] = acc[r] * scale;
  }
  __syncthreads();

  // v, transposed to [d][s] (B-operand layout for attn@v); pad cols zeroed
  for (int i = tid; i < 96 * 224; i += 128) {
    int d = i / 224, s = i - d * 224;
    bf16 val = (bf16)0.f;
    if (s < S) val = v[((size_t)b * S + s) * 768 + hh * 96 + d];
    kv_lds[d * 224 + s] = val;
  }
  // masked softmax, one thread per query row
  if (tid < 64) {
    int row = tid;
    float mx = -1e30f;
    for (int s = 0; s < S; s++) mx = fmaxf(mx, sc[row * 208 + s]);
    float sum = 0.f;
    for (int s = 0; s < S; s++) { float e = __expf(sc[row * 208 + s] - mx); sc[row * 208 + s] = e; sum += e; }
    float inv = 1.0f / sum;
    for (int s = 0; s < S; s++)   at[row * 224 + s] = (bf16)(sc[row * 208 + s] * inv);
    for (int s = S; s < 224; s++) at[row * 224 + s] = (bf16)0.f;
  }
  __syncthreads();

  for (int dt = 0; dt < 6; ++dt) {
    v8f acc = {};
#pragma unroll
    for (int kk = 0; kk < 7; ++kk) {
      v16bf a  = frag_a(at, 224, m, kk * 32, lane);
      v16bf bb = frag_b(kv_lds, 224, dt * 16, kk * 32, lane);
      acc = WMMA_BF16(a, bb, acc);
    }
#pragma unroll
    for (int r = 0; r < 8; r++) {
      int c = c0 + m + r + 8 * lh;
      if (c < 1000)
        h_att[((size_t)b * 1000 + c) * 768 + hh * 96 + dt * 16 + ln] = (bf16)acc[r];
    }
  }
}

// ---------------- grouped per-class projection + logits ----------------
// Per class c: (64x768) @ (768x32pad); cols 0..19 -> h_proj, col 20 -> logits.
__global__ __launch_bounds__(128)
void group_proj(const bf16* __restrict__ hT, const bf16* __restrict__ protoT,
                const float* __restrict__ proto_bias, const float* __restrict__ pool_bias,
                const float* __restrict__ temperature,
                float* __restrict__ hproj_out, float* __restrict__ logits_out) {
  const int c = blockIdx.x;
  const int lane = threadIdx.x & 31, wave = threadIdx.x >> 5;
  const int ln = lane & 15, lh = lane >> 4;
  const bf16* Ab = hT + (size_t)c * 64 * 768;
  const bf16* Bb = protoT + (size_t)c * 32 * 768;
  v8f acc0 = {}, acc1 = {};
  for (int k0 = 0; k0 < 768; k0 += 32) {
    v16bf a  = frag_a(Ab, 768, wave * 16, k0, lane);
    v16bf b0 = frag_b(Bb, 768, 0,  k0, lane);
    v16bf b1 = frag_b(Bb, 768, 16, k0, lane);
    acc0 = WMMA_BF16(a, b0, acc0);
    acc1 = WMMA_BF16(a, b1, acc1);
  }
  const float temp = temperature[0];
#pragma unroll
  for (int r = 0; r < 8; r++) {
    int b = wave * 16 + r + 8 * lh;
    hproj_out[((size_t)b * 1000 + c) * 20 + ln] = acc0[r] + proto_bias[c * 20 + ln];
    int p1 = 16 + ln;
    if (p1 < 20)
      hproj_out[((size_t)b * 1000 + c) * 20 + p1] = acc1[r] + proto_bias[c * 20 + p1];
    else if (p1 == 20)
      logits_out[(size_t)b * 1000 + c] = temp * (acc1[r] + pool_bias[c]);
  }
}

// ---------------- host launch ----------------
extern "C" void kernel_launch(void* const* d_in, const int* in_sizes, int n_in,
                              void* d_out, int out_size, void* d_ws, size_t ws_size,
                              hipStream_t stream) {
  const float* x        = (const float*)d_in[0];
  const float* W_embed  = (const float*)d_in[1];
  const float* b_embed  = (const float*)d_in[2];
  const float* query    = (const float*)d_in[3];
  const float* in_w     = (const float*)d_in[4];
  const float* in_b     = (const float*)d_in[5];
  const float* out_w    = (const float*)d_in[6];
  const float* out_b    = (const float*)d_in[7];
  const float* temp     = (const float*)d_in[8];
  const float* pool     = (const float*)d_in[9];
  const float* pool_b   = (const float*)d_in[10];
  const float* proto    = (const float*)d_in[11];
  const float* proto_b  = (const float*)d_in[12];

  const int B = 64, S = 196, F = 2048, E = 768, C = 1000;
  const int MS = B * S;   // 12544
  const int MC = B * C;   // 64000

  char* w = (char*)d_ws;
  size_t off = 0;
  auto alloc = [&](size_t elems) { bf16* p = (bf16*)(w + off); off += ((elems * 2 + 255) & ~(size_t)255); return p; };
  bf16* xb     = alloc((size_t)MS * F);
  bf16* wet    = alloc((size_t)E * F);        // W_embed^T
  bf16* ipw    = alloc((size_t)3 * E * E);    // in_proj_w ([q;k;v] rows, NK layout)
  bf16* opw    = alloc((size_t)E * E);
  bf16* qeb    = alloc((size_t)C * E);
  bf16* e_b    = alloc((size_t)MS * E);
  bf16* k_b    = alloc((size_t)MS * E);
  bf16* v_b    = alloc((size_t)MS * E);
  bf16* q_b    = alloc((size_t)C * E);
  bf16* hatt   = alloc((size_t)MC * E);
  bf16* hT     = alloc((size_t)MC * E);       // [c][b][E]
  bf16* protoT = alloc((size_t)C * 32 * E);

  float* out_logits = (float*)d_out;
  float* out_h      = (float*)d_out + 64000;
  float* out_query  = (float*)d_out + 64000 + 49152000;
  float* out_hproj  = (float*)d_out + 64000 + 49152000 + 768000;

  auto blks = [](int n) { return (n + 255) / 256; };
  cvt_f32_bf16<<<blks(MS * F), 256, 0, stream>>>(x, xb, MS * F);
  cvt_f32_bf16<<<blks(3 * E * E), 256, 0, stream>>>(in_w, ipw, 3 * E * E);
  cvt_f32_bf16<<<blks(E * E), 256, 0, stream>>>(out_w, opw, E * E);
  cvt_f32_bf16<<<blks(C * E), 256, 0, stream>>>(query, qeb, C * E);
  transpose_wembed<<<blks(E * F), 256, 0, stream>>>(W_embed, wet);
  pack_proto<<<blks(C * 32 * E), 256, 0, stream>>>(proto, pool, protoT);
  copy_f32<<<blks(C * E), 256, 0, stream>>>(query, out_query, C * E);

  // e = relu(x @ W_embed + b): M=12544 N=768 K=2048
  gemm_bf16<<<dim3(E / 64, (MS + 63) / 64), 128, 0, stream>>>(
      xb, wet, b_embed, nullptr, e_b, 0, MS, E, F, 1);
  // k = e @ wk^T + bk ; v = e @ wv^T + bv
  gemm_bf16<<<dim3(E / 64, (MS + 63) / 64), 128, 0, stream>>>(
      e_b, ipw + (size_t)E * E, in_b + E, nullptr, k_b, 0, MS, E, E, 0);
  gemm_bf16<<<dim3(E / 64, (MS + 63) / 64), 128, 0, stream>>>(
      e_b, ipw + (size_t)2 * E * E, in_b + 2 * E, nullptr, v_b, 0, MS, E, E, 0);
  // q = query @ wq^T + bq: M=1000
  gemm_bf16<<<dim3(E / 64, (C + 63) / 64), 128, 0, stream>>>(
      qeb, ipw, in_b, nullptr, q_b, 0, C, E, E, 0);
  // attention: (c-chunk, head, batch)
  attn_kernel<<<dim3((C + 63) / 64, 8, B), 128, 0, stream>>>(q_b, k_b, v_b, hatt);
  // out_proj + leaky; f32 h to d_out, bf16 h to [c][b][E] layout
  gemm_bf16<<<dim3(E / 64, MC / 64), 128, 0, stream>>>(
      hatt, opw, out_b, out_h, hT, C, MC, E, E, 2);
  // grouped proto/pooling: h_proj + logits
  group_proj<<<dim3(C), 128, 0, stream>>>(hT, protoT, proto_b, pool_b, temp,
                                          out_hproj, out_logits);
}
